// MultiHeadAttention_10479720202500
// MI455X (gfx1250) — matrix operand order repaired and checked by reading, not compiled, not run on hardware
//
#include <hip/hip_runtime.h>

// ---------------------------------------------------------------------------
// MultiHeadAttention forward for gfx1250 (MI455X), wave32 + WMMA f16 16x16x32.
// B=2, L=2048, HIDDEN=1024, HEADS=16, D_HEAD=64, SCALE=8.
// Flash attention; K/V 64-key blocks staged into LDS once per workgroup by the
// Tensor Data Mover (tensor_load_to_lds, TENSORcnt), double-buffered so DMA
// overlaps WMMA compute. Softmax needs only 2 xor-16 permutes per block.
// ---------------------------------------------------------------------------

#define B_    2
#define L_    2048
#define HID_  1024
#define H_    16
#define DH_   64
#define M_    (B_ * L_)          // 4096 rows for projection GEMMs

typedef __attribute__((ext_vector_type(16))) _Float16 v16h;
typedef __attribute__((ext_vector_type(8)))  _Float16 v8h;
typedef __attribute__((ext_vector_type(8)))  float    v8f;
typedef __attribute__((ext_vector_type(4)))  unsigned v4u;
typedef __attribute__((ext_vector_type(8)))  int      v8i;
typedef __attribute__((ext_vector_type(4)))  int      v4i;

// ---------------- fragment loaders (wave32 layouts per CDNA5 ISA 7.12.2) ----

// A-matrix 16x32 f16, row-major source with row stride ld (in halfs).
__device__ __forceinline__ v16h load_a_frag(const _Float16* p, int ld) {
    const int lane = threadIdx.x & 31;
    const int row  = lane & 15;
    const int kb   = (lane < 16) ? 0 : 8;
    const _Float16* r0 = p + row * ld + kb;
    v8h lo = *(const v8h*)(r0);
    v8h hi = *(const v8h*)(r0 + 16);
    v16h out;
#pragma unroll
    for (int j = 0; j < 8; ++j) { out[j] = lo[j]; out[j + 8] = hi[j]; }
    return out;
}

// B-matrix 32x16 f16 from a row-major TRANSPOSED buffer Bt[n][k], stride ld.
__device__ __forceinline__ v16h load_bt_frag(const _Float16* bt, int ld) {
    const int lane = threadIdx.x & 31;
    const int col  = lane & 15;
    const int kb   = (lane < 16) ? 0 : 8;
    const _Float16* r0 = bt + col * ld + kb;
    v8h lo = *(const v8h*)(r0);
    v8h hi = *(const v8h*)(r0 + 16);
    v16h out;
#pragma unroll
    for (int j = 0; j < 8; ++j) { out[j] = lo[j]; out[j + 8] = hi[j]; }
    return out;
}

// ---------------- TDM: 64x64 f16 2-D tile -> LDS (6-arg builtin) -----------
// D# per CDNA5 ISA ch.8: group0 = {flags, lds_addr, global_addr, type=2},
// group1 = {data_size=2B, tensor dims 64x64, tile 64x64, dim0 stride}.
// Groups 2/3 + trailing group zero-filled (2-D tensor: tile_dim2 = 0).
__device__ __forceinline__ void tdm_load_tile_64x64(unsigned lds_off,
                                                    const _Float16* gptr,
                                                    unsigned row_stride_elems) {
    const unsigned long long ga = (unsigned long long)(uintptr_t)gptr;
    v4u g0;
    g0[0] = 1u;                                         // count=1, user mode
    g0[1] = lds_off;                                    // LDS byte address
    g0[2] = (unsigned)ga;                               // global addr lo32
    g0[3] = ((unsigned)(ga >> 32) & 0x01FFFFFFu)        // global addr hi25
            | (2u << 30);                               // type = 2 ("image")
    v8i g1;
    g1[0] = 0x00010000;                                 // data_size=2B
    g1[1] = (int)(64u << 16);                           // tensor_dim0 = 64
    g1[2] = (int)(64u << 16);                           // tensor_dim1 = 64
    g1[3] = (int)(64u << 16);                           // tile_dim0 = 64
    g1[4] = 64;                                         // tile_dim1=64, dim2=0
    g1[5] = (int)row_stride_elems;                      // tensor_dim0_stride
    g1[6] = 0;
    g1[7] = 0;
    const v4i z4 = {0, 0, 0, 0};
    const v8i z8 = {0, 0, 0, 0, 0, 0, 0, 0};
    __builtin_amdgcn_tensor_load_to_lds(g0, g1, z4, z4, z8, 0);
}

// ---------------- conversion kernels ---------------------------------------

__global__ void cvt_f32_to_f16(const float* __restrict__ in,
                               _Float16* __restrict__ out, int n) {
    int i = blockIdx.x * blockDim.x + threadIdx.x;
    if (i < n) out[i] = (_Float16)in[i];
}

// Wt[n][k] = (f16) W[k][n]   (K = N = 1024)
__global__ void cvt_transpose_w(const float* __restrict__ w,
                                _Float16* __restrict__ wt) {
    int i = blockIdx.x * blockDim.x + threadIdx.x;   // i = k*1024 + n
    int k = i >> 10;
    int n = i & 1023;
    wt[n * HID_ + k] = (_Float16)w[i];
}

// ---------------- WMMA GEMM: out = A[M,K] @ Bt^T + bias --------------------
// Each wave computes a 16x64 tile (one A-fragment feeds 4 WMMAs per k-step).
// MODE 0: f16 out [B,H,L,DH], scaled (Q/K)   MODE 1: f16 out [B,H,DH,L] (V^T)
// MODE 2: f32 out row-major [M,N] (final projection)
template <int MODE>
__global__ void gemm_wmma(const _Float16* __restrict__ A,
                          const _Float16* __restrict__ Bt,
                          const float* __restrict__ bias,
                          _Float16* __restrict__ outH,
                          float* __restrict__ outF,
                          float scale) {
    const int lane  = threadIdx.x & 31;
    const int wave  = threadIdx.x >> 5;
    const int tile  = blockIdx.x * 8 + wave;     // 4096 wave-tiles total
    const int mtile = tile >> 4;                 // 256 m-tiles
    const int nt4   = tile & 15;                 // 16 n-groups of 64

    const _Float16* ap = A  + (size_t)mtile * 16 * HID_;
    const _Float16* bp = Bt + (size_t)nt4 * 64 * HID_;

    v8f c[4] = {};
#pragma unroll 2
    for (int k0 = 0; k0 < HID_; k0 += 32) {
        v16h a = load_a_frag(ap + k0, HID_);
#pragma unroll
        for (int s = 0; s < 4; ++s) {
            v16h b = load_bt_frag(bp + (size_t)s * 16 * HID_ + k0, HID_);
            c[s] = __builtin_amdgcn_wmma_f32_16x16x32_f16(false, a, false, b,
                                                          (short)0, c[s],
                                                          false, false);
        }
    }

    const int col   = lane & 15;
    const int rbase = (lane < 16) ? 0 : 8;

#pragma unroll
    for (int s = 0; s < 4; ++s) {
        const int n    = nt4 * 64 + s * 16 + col;
        const float bv = bias[n];
#pragma unroll
        for (int r = 0; r < 8; ++r) {
            const int row = mtile * 16 + rbase + r;  // global m
            const float val = c[s][r] + bv;
            if (MODE == 2) {
                outF[(size_t)row * HID_ + n] = val;
            } else {
                const int bb = row >> 11;            // / L
                const int l  = row & (L_ - 1);
                const int h  = n >> 6;               // / DH
                const int d  = n & (DH_ - 1);
                if (MODE == 0) {
                    outH[((((size_t)bb * H_ + h) * L_ + l) << 6) + d] =
                        (_Float16)(val * scale);
                } else { // MODE 1: V transposed [B,H,DH,L]
                    outH[(((size_t)bb * H_ + h) * DH_ + d) * L_ + l] =
                        (_Float16)val;
                }
            }
        }
    }
}

// ---------------- flash attention ------------------------------------------
// One workgroup = 8 waves = 8 consecutive 16-query tiles of ONE (b,h); K/V
// 64-key blocks are TDM-staged into LDS once per WG and shared by all waves.
__global__ void flash_attn(const _Float16* __restrict__ Q,
                           const _Float16* __restrict__ Kk,
                           const _Float16* __restrict__ Vt,
                           const float* __restrict__ mask,
                           _Float16* __restrict__ ctx) {
    __shared__ _Float16 kbuf[2][64 * 64];        // [key][d]   8KB each
    __shared__ _Float16 vbuf[2][64 * 64];        // [d][key]   8KB each
    __shared__ float    sbuf[8][16 * 64];        // per-wave S tiles, 32KB

    const int lane = threadIdx.x & 31;
    const int wave = threadIdx.x >> 5;
    const int bh   = blockIdx.x >> 4;            // 32 (b,h) pairs
    const int qt   = (blockIdx.x & 15) * 8 + wave;
    const int h    = bh & (H_ - 1);
    const int bb   = bh >> 4;

    const int col   = lane & 15;
    const int rbase = (lane < 16) ? 0 : 8;       // C-layout row base
    const int arow  = lane & 15;                 // A-layout row owned by lane
    const int kb8   = (lane < 16) ? 0 : 8;       // A-layout K sub-base

    const size_t headQ = (((size_t)bb * H_ + h) * L_) * DH_;
    const _Float16* qp = Q + headQ + (size_t)qt * 16 * DH_;
    const _Float16* kg = Kk + headQ;                       // K [L,DH]
    const _Float16* vg = Vt + (((size_t)bb * H_ + h) * DH_) * L_;  // V^T [DH,L]
    const float*    mk = mask + (size_t)bb * L_;

    const v16h a0 = load_a_frag(qp, DH_);
    const v16h a1 = load_a_frag(qp + 32, DH_);

    v8f o[4] = {};
    float mrun = -3.0e30f;   // stats for row `arow` (replicated in both halves)
    float lrun = 0.0f;

    float* sw = &sbuf[wave][0];
    const float* srow = sw + arow * 64;

    const unsigned koff[2] = { (unsigned)(uintptr_t)&kbuf[0][0],
                               (unsigned)(uintptr_t)&kbuf[1][0] };
    const unsigned voff[2] = { (unsigned)(uintptr_t)&vbuf[0][0],
                               (unsigned)(uintptr_t)&vbuf[1][0] };

    // prologue: DMA first K/V block into buffer 0
    if (wave == 0) {
        tdm_load_tile_64x64(koff[0], kg, DH_);
        tdm_load_tile_64x64(voff[0], vg, L_);
    }

    int cur = 0;
    for (int k0 = 0; k0 < L_; k0 += 64) {
        if (wave == 0) __builtin_amdgcn_s_wait_tensorcnt((short)0);
        __syncthreads();                         // staged data visible to WG
        if (wave == 0 && (k0 + 64) < L_) {       // prefetch next block (DMA
            tdm_load_tile_64x64(koff[cur ^ 1], kg + (size_t)(k0 + 64) * DH_, DH_);
            tdm_load_tile_64x64(voff[cur ^ 1], vg + (k0 + 64), L_);
        }                                        //  overlaps compute below)
        const _Float16* kp = &kbuf[cur][0];
        const _Float16* vp = &vbuf[cur][0];

        // ---- S = Q @ K^T (16x64) : 8 WMMAs, K fragments from LDS ----
        v8f s[4];
#pragma unroll
        for (int kc = 0; kc < 4; ++kc) {
            v16h blo = load_bt_frag(kp + kc * 16 * 64, 64);
            v16h bhi = load_bt_frag(kp + kc * 16 * 64 + 32, 64);
            v8f z = {};
            z = __builtin_amdgcn_wmma_f32_16x16x32_f16(false, a0, false, blo,
                                                       (short)0, z, false, false);
            s[kc] = __builtin_amdgcn_wmma_f32_16x16x32_f16(false, a1, false, bhi,
                                                           (short)0, z, false, false);
        }

        // ---- mask + C->A transpose of S through per-wave LDS tile ----
#pragma unroll
        for (int kc = 0; kc < 4; ++kc) {
            const float pen = mk[k0 + kc * 16 + col] * -10000.0f;
#pragma unroll
            for (int r = 0; r < 8; ++r)
                sw[(rbase + r) * 64 + kc * 16 + col] = s[kc][r] + pen;
        }
        asm volatile("s_wait_dscnt 0x0" ::: "memory");

        // lane reads ITS OWN row's A-fragment positions (4 runs of 8 f32)
        v8f g0, g1, g2, g3;
#pragma unroll
        for (int j = 0; j < 8; ++j) {
            g0[j] = srow[kb8 + j];
            g1[j] = srow[16 + kb8 + j];
            g2[j] = srow[32 + kb8 + j];
            g3[j] = srow[48 + kb8 + j];
        }

        // ---- online softmax: per-row stats, 2 xor-16 permutes total ----
        float lm = -3.0e30f;
#pragma unroll
        for (int j = 0; j < 8; ++j)
            lm = fmaxf(lm, fmaxf(fmaxf(g0[j], g1[j]), fmaxf(g2[j], g3[j])));
        lm = fmaxf(lm, __shfl_xor(lm, 16, 32));          // full-row max
        const float mnew = fmaxf(mrun, lm);
        const float corr = __expf(mrun - mnew);
        mrun = mnew;

        v16h pf0, pf1;                                   // P as A-fragments
        float ls = 0.0f;
#pragma unroll
        for (int j = 0; j < 8; ++j) {
            float p0 = __expf(g0[j] - mnew);
            float p1 = __expf(g1[j] - mnew);
            float p2 = __expf(g2[j] - mnew);
            float p3 = __expf(g3[j] - mnew);
            ls += (p0 + p1) + (p2 + p3);
            pf0[j] = (_Float16)p0; pf0[j + 8] = (_Float16)p1;
            pf1[j] = (_Float16)p2; pf1[j + 8] = (_Float16)p3;
        }
        ls += __shfl_xor(ls, 16, 32);                    // full-row sum
        lrun = lrun * corr + ls;

        // broadcast correction into C layout (row = rbase + r)
#pragma unroll
        for (int r = 0; r < 8; ++r) {
            const float cc = __shfl(corr, rbase + r, 32);
            o[0][r] *= cc; o[1][r] *= cc; o[2][r] *= cc; o[3][r] *= cc;
        }

        // ---- O += P @ V : 8 WMMAs, V fragments from LDS ----
#pragma unroll
        for (int c = 0; c < 4; ++c) {
            v16h vlo = load_bt_frag(vp + c * 16 * 64, 64);
            v16h vhi = load_bt_frag(vp + c * 16 * 64 + 32, 64);
            o[c] = __builtin_amdgcn_wmma_f32_16x16x32_f16(false, pf0, false, vlo,
                                                          (short)0, o[c],
                                                          false, false);
            o[c] = __builtin_amdgcn_wmma_f32_16x16x32_f16(false, pf1, false, vhi,
                                                          (short)0, o[c],
                                                          false, false);
        }
        cur ^= 1;
    }

    // ---- normalize and write context [B, L, HID] f16 ----
    const float inv = 1.0f / lrun;                       // stats layout
#pragma unroll
    for (int r = 0; r < 8; ++r) {
        const float ic = __shfl(inv, rbase + r, 32);     // to C layout
        const int l = qt * 16 + rbase + r;
        _Float16* cp = ctx + ((size_t)bb * L_ + l) * HID_ + h * DH_ + col;
        cp[0]  = (_Float16)(o[0][r] * ic);
        cp[16] = (_Float16)(o[1][r] * ic);
        cp[32] = (_Float16)(o[2][r] * ic);
        cp[48] = (_Float16)(o[3][r] * ic);
    }
}

// ---------------- host-side orchestration ----------------------------------

extern "C" void kernel_launch(void* const* d_in, const int* in_sizes, int n_in,
                              void* d_out, int out_size, void* d_ws, size_t ws_size,
                              hipStream_t stream) {
    (void)in_sizes; (void)n_in; (void)out_size; (void)ws_size;

    const float* x    = (const float*)d_in[0];
    const float* mask = (const float*)d_in[1];
    const float* Wq   = (const float*)d_in[2];
    const float* bq   = (const float*)d_in[3];
    const float* Wk   = (const float*)d_in[4];
    const float* bk   = (const float*)d_in[5];
    const float* Wv   = (const float*)d_in[6];
    const float* bv   = (const float*)d_in[7];
    const float* Wo   = (const float*)d_in[8];
    const float* bo   = (const float*)d_in[9];
    float* out = (float*)d_out;

    // workspace carve-up (halfs)
    _Float16* ws = (_Float16*)d_ws;
    const size_t NXH = (size_t)M_ * HID_;        // 4194304
    const size_t NW  = (size_t)HID_ * HID_;      // 1048576
    _Float16* xh  = ws;                          // x in f16
    _Float16* Wqt = xh  + NXH;                   // transposed f16 weights
    _Float16* Wkt = Wqt + NW;
    _Float16* Wvt = Wkt + NW;
    _Float16* Wot = Wvt + NW;
    _Float16* Qb  = Wot + NW;                    // [B,H,L,DH], pre-scaled
    _Float16* Kb  = Qb  + NXH;                   // [B,H,L,DH]
    _Float16* Vtb = Kb  + NXH;                   // [B,H,DH,L]
    _Float16* ctx = Vtb + NXH;                   // [B,L,HID]

    const float inv_scale = 0.125f;              // 1/sqrt(64)

    // 1) precision conversion + weight transposes
    cvt_f32_to_f16<<<(int)(NXH / 256), 256, 0, stream>>>(x, xh, (int)NXH);
    cvt_transpose_w<<<(int)(NW / 256), 256, 0, stream>>>(Wq, Wqt);
    cvt_transpose_w<<<(int)(NW / 256), 256, 0, stream>>>(Wk, Wkt);
    cvt_transpose_w<<<(int)(NW / 256), 256, 0, stream>>>(Wv, Wvt);
    cvt_transpose_w<<<(int)(NW / 256), 256, 0, stream>>>(Wo, Wot);

    // 2) Q/K/V projections (4096 wave-tiles -> 512 blocks of 8 waves)
    gemm_wmma<0><<<512, 256, 0, stream>>>(xh, Wqt, bq, Qb,  nullptr, inv_scale);
    gemm_wmma<0><<<512, 256, 0, stream>>>(xh, Wkt, bk, Kb,  nullptr, 1.0f);
    gemm_wmma<1><<<512, 256, 0, stream>>>(xh, Wvt, bv, Vtb, nullptr, 1.0f);

    // 3) flash attention (32 heads x 16 query-groups = 512 blocks)
    flash_attn<<<512, 256, 0, stream>>>(Qb, Kb, Vtb, mask, ctx);

    // 4) output projection -> f32 d_out
    gemm_wmma<2><<<512, 256, 0, stream>>>(ctx, Wot, bo, nullptr, out, 1.0f);
}
